// FirDownsample2D_16698832847479
// MI455X (gfx1250) — matrix operand order repaired and checked by reading, not compiled, not run on hardware
//
#include <hip/hip_runtime.h>

// ---------------------------------------------------------------------------
// FirDownsample2D for MI455X (gfx1250, wave32, WMMA)
//   stage 1: weight f32 -> bf16 [cout][k], k-offset table  (k = ci*9+kh*3+kw)
//   stage 2: depthwise 4x4 FIR (1,3,3,1 outer, /64) -> y bf16 [16][256][129][129]
//   stage 3: implicit GEMM, v_wmma_f32_16x16x32_bf16, double-buffered LDS,
//            2x2 WMMA tile per wave, f32 accum, + bias
// ---------------------------------------------------------------------------

typedef __attribute__((ext_vector_type(16))) __bf16        v16bf;
typedef __attribute__((ext_vector_type(8)))  float         v8f;
typedef __attribute__((ext_vector_type(4)))  unsigned int  uint4e;

union Frag16 {            // 32 bytes: one bf16 WMMA A/B fragment per lane
    v16bf  v;
    uint4e q[2];
};

#define XN   16
#define XC   256
#define XH   128
#define XW   128
#define YH   129
#define YW   129
#define OH   64
#define OW   64
#define KTOT (256 * 9)        // 2304, multiple of 32
#define KCHUNK 32
#define TILE_PIX 32           // 2 N-tiles of 16 pixels per block

__device__ __forceinline__ unsigned short f32_to_bf16(float f) {
    unsigned int u = __float_as_uint(f);
    unsigned int r = u + 0x7FFFu + ((u >> 16) & 1u);   // round-to-nearest-even
    return (unsigned short)(r >> 16);
}

// ---------------------------------------------------------------------------
// Stage 1a: weight (256,256,3,3) f32 OIHW -> bf16 [cout][2304]
// ---------------------------------------------------------------------------
__global__ __launch_bounds__(256) void cvt_weight(const float* __restrict__ w,
                                                  unsigned short* __restrict__ wbf) {
    int i = blockIdx.x * 256 + threadIdx.x;
    if (i < 256 * KTOT) wbf[i] = f32_to_bf16(w[i]);
}

// Stage 1b: gather-offset table  koff[k] = ci*129*129 + kh*129 + kw
__global__ __launch_bounds__(256) void make_koff(int* __restrict__ koff) {
    int k = blockIdx.x * 256 + threadIdx.x;
    if (k < KTOT) {
        int ci = k / 9;
        int r9 = k - ci * 9;
        int kh = r9 / 3;
        int kw = r9 - kh * 3;
        koff[k] = ci * (YH * YW) + kh * YW + kw;
    }
}

// ---------------------------------------------------------------------------
// Stage 2: separable FIR, zero pad 2/2, output (16,256,129,129) bf16
// ---------------------------------------------------------------------------
__global__ __launch_bounds__(256) void fir_kernel(const float* __restrict__ x,
                                                  unsigned short* __restrict__ ybf,
                                                  int total) {
    int idx = blockIdx.x * 256 + threadIdx.x;
    if (idx >= total) return;
    int iw = idx % YW;
    int t  = idx / YW;
    int ih = t % YH;
    int nc = t / YH;                                  // fused (n,c) in [0,4096)
    const float* xb = x + (long long)nc * (XH * XW);

    const float f0 = 0.125f, f1 = 0.375f;
    const float fa[4] = {f0, f1, f1, f0};

    float s = 0.0f;
    #pragma unroll
    for (int a = 0; a < 4; ++a) {
        int xi = ih - 2 + a;
        if (xi < 0 || xi >= XH) continue;
        const float* row = xb + xi * XW;
        float rs = 0.0f;
        #pragma unroll
        for (int b = 0; b < 4; ++b) {
            int xj = iw - 2 + b;
            if (xj >= 0 && xj < XW) rs += fa[b] * row[xj];
        }
        s += fa[a] * rs;
    }
    ybf[idx] = f32_to_bf16(s);
}

// ---------------------------------------------------------------------------
// Stage 3: implicit GEMM with WMMA, double-buffered LDS pipeline.
//   Block: 32 output pixels x all 256 couts. 8 waves; wave w owns couts
//   [32w, 32w+32) as two 16-row A tiles, times two 16-pixel B tiles.
// ---------------------------------------------------------------------------
__global__ __launch_bounds__(256) void conv_wmma(const unsigned short* __restrict__ ybf,
                                                 const unsigned short* __restrict__ wbf,
                                                 const int* __restrict__ koff,
                                                 const float* __restrict__ bias,
                                                 float* __restrict__ out) {
    // [buf][ntile][lane*16 + half] : 2 x 2 x 512 bf16 = 4 KB
    __shared__ __attribute__((aligned(64))) unsigned short lds[2 * 2 * 512];

    const int tid   = threadIdx.x;
    const int lane  = tid & 31;
    const int wave  = tid >> 5;                       // 0..7
    const int pbase = blockIdx.x * TILE_PIX;

    // ---- gather setup: thread stages pixels n = wave + 8t (t=0..3), kk = lane
    int ybase[4];
    int dstoff[4];
    #pragma unroll
    for (int t = 0; t < 4; ++t) {
        int n   = wave + 8 * t;
        int pix = pbase + n;
        int ow = pix & 63, oh = (pix >> 6) & 63, nb = pix >> 12;
        ybase[t] = nb * (XC * YH * YW) + (2 * oh) * YW + 2 * ow;
        // B layout (bf16 32x16): lanes 0-15 hold K=0..7,16..23, lanes 16-31 K=8..15,24..31
        int laned = (n & 15) + ((lane & 8) << 1);
        dstoff[t] = (n >> 4) * 512 + laned * 16 + (lane & 7) + ((lane & 16) >> 1);
    }

    // ---- A fragment addressing (bf16 16x32 per ISA table):
    //   lanes 0-15 : halves 0-7 = K kb+0..7,  halves 8-15 = K kb+16..23
    //   lanes 16-31: halves 0-7 = K kb+8..15, halves 8-15 = K kb+24..31
    const int hi   = lane >> 4;
    const int mrow = lane & 15;
    const int row0 = (wave * 32 + mrow) * KTOT;       // cout tile 0
    const int row1 = row0 + 16 * KTOT;                // cout tile 1

    v8f acc00 = {}, acc01 = {}, acc10 = {}, acc11 = {};

    // ---- prologue: gather chunk 0 into buffer 0
    {
        int ko = koff[lane];
        #pragma unroll
        for (int t = 0; t < 4; ++t) lds[dstoff[t]] = ybf[ybase[t] + ko];
    }
    __syncthreads();

    int buf = 0;
    #pragma unroll 2
    for (int kb = 0; kb < KTOT; kb += KCHUNK) {
        // ---- gather next chunk into registers (overlaps WMMA below);
        //      clamp tail to keep the pipeline branchless
        int kbn = kb + KCHUNK;
        if (kbn >= KTOT) kbn = 0;
        int ko = koff[kbn + lane];
        unsigned short nv0 = ybf[ybase[0] + ko];
        unsigned short nv1 = ybf[ybase[1] + ko];
        unsigned short nv2 = ybf[ybase[2] + ko];
        unsigned short nv3 = ybf[ybase[3] + ko];

        // ---- fragments from current buffer
        const unsigned short* cb = lds + buf * 1024;
        const int o = kb + hi * 8;
        Frag16 a0, a1, b0, b1;
        a0.q[0] = *(const uint4e*)(wbf + row0 + o);
        a0.q[1] = *(const uint4e*)(wbf + row0 + o + 16);
        a1.q[0] = *(const uint4e*)(wbf + row1 + o);
        a1.q[1] = *(const uint4e*)(wbf + row1 + o + 16);
        b0.q[0] = *(const uint4e*)(cb + lane * 16);
        b0.q[1] = *(const uint4e*)(cb + lane * 16 + 8);
        b1.q[0] = *(const uint4e*)(cb + 512 + lane * 16);
        b1.q[1] = *(const uint4e*)(cb + 512 + lane * 16 + 8);

        __builtin_prefetch(wbf + row0 + o + 2 * KCHUNK, 0, 1);

        acc00 = __builtin_amdgcn_wmma_f32_16x16x32_bf16(
            false, a0.v, false, b0.v, (short)0, acc00, false, false);
        acc01 = __builtin_amdgcn_wmma_f32_16x16x32_bf16(
            false, a0.v, false, b1.v, (short)0, acc01, false, false);
        acc10 = __builtin_amdgcn_wmma_f32_16x16x32_bf16(
            false, a1.v, false, b0.v, (short)0, acc10, false, false);
        acc11 = __builtin_amdgcn_wmma_f32_16x16x32_bf16(
            false, a1.v, false, b1.v, (short)0, acc11, false, false);

        // ---- stage next chunk into the other buffer
        unsigned short* nbuf = lds + (buf ^ 1) * 1024;
        nbuf[dstoff[0]] = nv0;
        nbuf[dstoff[1]] = nv1;
        nbuf[dstoff[2]] = nv2;
        nbuf[dstoff[3]] = nv3;

        __syncthreads();                              // single barrier per chunk
        buf ^= 1;
    }

    // ---- epilogue: C/D layout (VGPR r -> M = r + 8*hi, N = lane&15) ----
    const int col  = lane & 15;
    const int mofs = hi * 8;
    #pragma unroll
    for (int j = 0; j < 2; ++j) {
        int pix = pbase + j * 16 + col;
        int ow = pix & 63, oh = (pix >> 6) & 63, nb = pix >> 12;
        int obase = (nb * 256) * (OH * OW) + oh * OW + ow;
        v8f accA = j ? acc01 : acc00;                 // cout tile 0
        v8f accB = j ? acc11 : acc10;                 // cout tile 1
        #pragma unroll
        for (int r = 0; r < 8; ++r) {
            int co0 = wave * 32 + mofs + r;
            int co1 = co0 + 16;
            out[obase + co0 * (OH * OW)] = accA[r] + bias[co0];
            out[obase + co1 * (OH * OW)] = accB[r] + bias[co1];
        }
    }
}

// ---------------------------------------------------------------------------
extern "C" void kernel_launch(void* const* d_in, const int* in_sizes, int n_in,
                              void* d_out, int out_size, void* d_ws, size_t ws_size,
                              hipStream_t stream) {
    const float* x    = (const float*)d_in[0];
    const float* w    = (const float*)d_in[1];
    const float* bias = (const float*)d_in[2];
    float* out        = (float*)d_out;

    // workspace layout: bf16 weight | koff table | bf16 y
    unsigned short* wbf  = (unsigned short*)d_ws;
    int*            koff = (int*)((char*)d_ws + (size_t)256 * KTOT * 2);
    unsigned short* ybf  = (unsigned short*)((char*)d_ws + (size_t)256 * KTOT * 2
                                                         + (size_t)KTOT * 4);

    cvt_weight<<<(256 * KTOT + 255) / 256, 256, 0, stream>>>(w, wbf);
    make_koff<<<(KTOT + 255) / 256, 256, 0, stream>>>(koff);

    const int ytotal = XN * XC * YH * YW;
    fir_kernel<<<(ytotal + 255) / 256, 256, 0, stream>>>(x, ybf, ytotal);

    const int npix = XN * OH * OW;                    // 65536
    conv_wmma<<<npix / TILE_PIX, 256, 0, stream>>>(ybf, wbf, koff, bias, out);
}
